// LSTM_31215822307545
// MI455X (gfx1250) — matrix operand order repaired and checked by reading, not compiled, not run on hardware
//
#include <hip/hip_runtime.h>
#include <hip/hip_bf16.h>

typedef __attribute__((ext_vector_type(16))) _Float16 v16h;
typedef __attribute__((ext_vector_type(2)))  _Float16 v2h;
typedef __attribute__((ext_vector_type(8)))  float    v8f;

#define LSTM_H 16
#define CHUNK 8   // timesteps per x-load chunk (16 floats = 4x b128)

#if __has_builtin(__builtin_amdgcn_tanhf)
#define TANHF(v) __builtin_amdgcn_tanhf(v)       // gfx1250 v_tanh_f32
#else
#define TANHF(v) (2.0f * __builtin_amdgcn_rcpf(1.0f + __expf(-2.0f * (v))) - 1.0f)
#endif

__device__ __forceinline__ float fast_sigmoid(float x) {
    return __builtin_fmaf(0.5f, TANHF(0.5f * x), 0.5f);   // mul + tanh + fma
}

// Pack two f32 into one b32 holding two f16 (v_cvt_pk_f16_f32).
__device__ __forceinline__ int pack2h(float a, float b) {
    v2h p; p.x = (_Float16)a; p.y = (_Float16)b;
    return __builtin_bit_cast(int, p);
}

// One wave (32 lanes) handles a 16-row batch tile for all T timesteps.
// State layout matches the WMMA D layout of the transposed formulation
// gates^T = [W_hh | W_ih] * [h ; x]^T :
//   lane%16 = batch row n within tile, half = lane/16,
//   register r (0..7) = hidden index j = r + 8*half.
__global__ __launch_bounds__(32)
void lstm_wmma_kernel(const float* __restrict__ x,
                      const float* __restrict__ W_ih,   // [64, 2]
                      const float* __restrict__ W_hh,   // [64, 16]
                      const float* __restrict__ b_ih,   // [64]
                      const float* __restrict__ b_hh,   // [64]
                      const float* __restrict__ W1,     // [8, 16]
                      const float* __restrict__ b1,     // [8]
                      const float* __restrict__ W2,     // [4, 8]
                      const float* __restrict__ b2,     // [4]
                      const float* __restrict__ W3,     // [1, 4]
                      const float* __restrict__ b3,     // [1]
                      float* __restrict__ out,          // [B]
                      int B, int T)
{
    const int lane = threadIdx.x & 31;
    const int half = lane >> 4;         // 0: lanes 0-15, 1: lanes 16-31
    const int l16  = lane & 15;
    const int b0   = blockIdx.x * 16;   // batch tile base

    // ---------------- Static A matrices (held in VGPRs for the whole loop)
    // A_g is 16x32 f16 (M = gate-within-group, K = 0..15 hidden, 16..17 input).
    // A layout (16-bit, 16x32): lanes 0-15 hold K=0..7 (elems 0-7) and K=16..23
    // (elems 8-15); lanes 16-31 hold K=8..15 and K=24..31.
    v16h A[4];
    #pragma unroll
    for (int g = 0; g < 4; ++g) {
        const int row = g * 16 + l16;               // gate index 0..63
        const float* wr = W_hh + row * LSTM_H;
        v16h a;
        #pragma unroll
        for (int e = 0; e < 8; ++e)
            a[e] = (_Float16)wr[half * 8 + e];      // K = 0..7 or 8..15
        #pragma unroll
        for (int e = 8; e < 16; ++e)
            a[e] = (_Float16)0.0f;                  // K = 16..23 / 24..31
        // lanes 0-15: K=16,17 carry the input-projection weights
        a[8] = (half == 0) ? (_Float16)W_ih[row * 2 + 0] : (_Float16)0.0f;
        a[9] = (half == 0) ? (_Float16)W_ih[row * 2 + 1] : (_Float16)0.0f;
        A[g] = a;
    }

    // Bias in D layout: C_g[r] = b[16g + r + 8*half] (independent of batch col)
    v8f Cb[4];
    #pragma unroll
    for (int g = 0; g < 4; ++g) {
        #pragma unroll
        for (int r = 0; r < 8; ++r) {
            const int gi = 16 * g + r + 8 * half;
            Cb[g][r] = b_ih[gi] + b_hh[gi];
        }
    }

    // Recurrent state in registers, D layout.
    float hs[8], cs[8];
    #pragma unroll
    for (int r = 0; r < 8; ++r) { hs[r] = 0.0f; cs[r] = 0.0f; }

    // Per-lane x stream: column n = l16 of this tile (clamped for safety).
    int brow = b0 + l16; if (brow > B - 1) brow = B - 1;
    const float* xrow = x + (size_t)brow * (size_t)T * 2u;

    // ---- one LSTM step, x values given in registers ----
    auto step = [&](float x0, float x1) {
        // Pack own h into f16 pairs, then one 4-dword cross-half exchange.
        int own[4], swp[4];
        #pragma unroll
        for (int j = 0; j < 4; ++j)
            own[j] = pack2h(hs[2 * j], hs[2 * j + 1]);
        #pragma unroll
        for (int j = 0; j < 4; ++j)
            swp[j] = __shfl(own[j], l16 + 16, 32);  // lanes<16 get upper half's h
        const int xpk = pack2h(x0, x1);

        // B matrix 32x16 f16: lanes 0-15 = h[0..7] (K=0..7), x_t (K=16,17);
        // lanes 16-31 = h[8..15] (K=8..15), rest zero.
        union { v16h h; int i[8]; } bu;
        bu.i[0] = own[0];
        bu.i[1] = own[1];
        bu.i[2] = own[2];
        bu.i[3] = own[3];
        #pragma unroll
        for (int j = 0; j < 4; ++j)
            bu.i[4 + j] = (half == 0 && j == 0) ? xpk : 0;
        const v16h Bm = bu.h;

        // gates^T for the whole tile: 4x v_wmma_f32_16x16x32_f16
        v8f Di = __builtin_amdgcn_wmma_f32_16x16x32_f16(false, A[0], false, Bm, (short)0, Cb[0], false, false);
        v8f Df = __builtin_amdgcn_wmma_f32_16x16x32_f16(false, A[1], false, Bm, (short)0, Cb[1], false, false);
        v8f Dg = __builtin_amdgcn_wmma_f32_16x16x32_f16(false, A[2], false, Bm, (short)0, Cb[2], false, false);
        v8f Do = __builtin_amdgcn_wmma_f32_16x16x32_f16(false, A[3], false, Bm, (short)0, Cb[3], false, false);

        // Elementwise LSTM cell update (f32, in registers).
        #pragma unroll
        for (int r = 0; r < 8; ++r) {
            const float ig = fast_sigmoid(Di[r]);
            const float fg = fast_sigmoid(Df[r]);
            const float gg = TANHF(Dg[r]);
            const float og = fast_sigmoid(Do[r]);
            cs[r] = fg * cs[r] + ig * gg;
            hs[r] = og * TANHF(cs[r]);
        }
    };

    // ---- chunked x streaming, ping-pong buffers, one chunk prefetch ahead ----
    const int nch = T / CHUNK;
    float xa[2 * CHUNK], xb[2 * CHUNK];

    auto load_chunk = [&](float* dst, int tbase) {
        const float4* p = (const float4*)(xrow + 2 * tbase);  // 64B aligned
        #pragma unroll
        for (int q = 0; q < CHUNK / 2; ++q) {
            float4 v = p[q];
            dst[4 * q + 0] = v.x; dst[4 * q + 1] = v.y;
            dst[4 * q + 2] = v.z; dst[4 * q + 3] = v.w;
        }
    };
    auto run_chunk = [&](const float* buf) {
        #pragma unroll
        for (int s = 0; s < CHUNK; ++s)
            step(buf[2 * s + 0], buf[2 * s + 1]);
    };

    if (nch > 0) load_chunk(xa, 0);
    int ch = 0;
    for (; ch + 1 < nch; ch += 2) {
        load_chunk(xb, (ch + 1) * CHUNK);   // prefetch odd chunk
        run_chunk(xa);
        if (ch + 2 < nch) load_chunk(xa, (ch + 2) * CHUNK);  // prefetch even
        run_chunk(xb);
    }
    if (ch < nch) run_chunk(xa);            // last unpaired chunk
    for (int t = nch * CHUNK; t < T; ++t)   // tail (T % CHUNK != 0)
        step(xrow[2 * t + 0], xrow[2 * t + 1]);

    // ---------------- Head: h @ W1^T + b1 -> @ W2^T + b2 -> @ W3^T + b3 -> sigmoid
    float y1[8];
    #pragma unroll
    for (int k = 0; k < 8; ++k) {
        const float* w1r = W1 + k * LSTM_H;
        float p = 0.0f;
        #pragma unroll
        for (int r = 0; r < 8; ++r)
            p += w1r[r + 8 * half] * hs[r];
        p += __shfl_xor(p, 16, 32);   // combine the two hidden halves
        y1[k] = p + b1[k];
    }
    float y2[4];
    #pragma unroll
    for (int k2 = 0; k2 < 4; ++k2) {
        float p = b2[k2];
        #pragma unroll
        for (int k = 0; k < 8; ++k)
            p += W2[k2 * 8 + k] * y1[k];
        y2[k2] = p;
    }
    float y3 = b3[0];
    #pragma unroll
    for (int k2 = 0; k2 < 4; ++k2)
        y3 += W3[k2] * y2[k2];

    const float o = fast_sigmoid(y3);
    if (half == 0 && (b0 + l16) < B)
        out[b0 + l16] = o;
}

extern "C" void kernel_launch(void* const* d_in, const int* in_sizes, int n_in,
                              void* d_out, int out_size, void* d_ws, size_t ws_size,
                              hipStream_t stream) {
    const float* x    = (const float*)d_in[0];
    const float* W_ih = (const float*)d_in[1];
    const float* W_hh = (const float*)d_in[2];
    const float* b_ih = (const float*)d_in[3];
    const float* b_hh = (const float*)d_in[4];
    const float* W1   = (const float*)d_in[5];
    const float* b1   = (const float*)d_in[6];
    const float* W2   = (const float*)d_in[7];
    const float* b2   = (const float*)d_in[8];
    const float* W3   = (const float*)d_in[9];
    const float* b3   = (const float*)d_in[10];
    float* out = (float*)d_out;

    const int B = out_size;                 // output is [B, 1]
    const int T = in_sizes[0] / (B * 2);    // x is [B, T, 2]

    const int tiles = (B + 15) / 16;        // one wave per 16-batch tile
    lstm_wmma_kernel<<<tiles, 32, 0, stream>>>(
        x, W_ih, W_hh, b_ih, b_hh, W1, b1, W2, b2, W3, b3, out, B, T);
}
